// STFDGCN_49417893708423
// MI455X (gfx1250) — matrix operand order repaired and checked by reading, not compiled, not run on hardware
//
#include <hip/hip_runtime.h>
#include <math.h>

// ---------------------------------------------------------------------------
// STFDGCN for MI455X (gfx1250, wave32).
// All K>=64 contractions go through v_wmma_f32_16x16x32_bf16 (bf16 A/B, f32 acc).
// Working set fits in 192MB L2 -> diffusion GEMMs are L2-resident; tiny ops
// (T<=13 attention, softmax, top-k, BN) stay on VALU.
// GEMM inner loop: uniform (scalar) segment pointer per K-tile, hoisted row
// addressing, branch-free clamped staging loads, LDS tiles stored in WMMA
// fragment order (wide ds_store_b128 on write, ds_load_b128 on read).
// ---------------------------------------------------------------------------

constexpr int kB   = 8;
constexpr int kT   = 12;
constexpr int kN   = 883;
constexpr int kCIN = 3;
constexpr int kC   = 64;
constexpr int kE   = 10;
constexpr int kHD  = 8;
constexpr int kTOPK = 48;
constexpr int kL   = 8;
constexpr int kT0  = 13;                       // time after left pad
constexpr long long RMAX = (long long)kB * kN * kT0;   // 91832 rows max

typedef __bf16 bf16;
typedef __attribute__((ext_vector_type(16))) __bf16 v16bf;
typedef __attribute__((ext_vector_type(8)))  __bf16 v8bf;
typedef __attribute__((ext_vector_type(8)))  float  v8f;

__device__ __forceinline__ bf16 f2bf(float f) {
  unsigned u = __builtin_bit_cast(unsigned, f);
  u += 0x7fffu + ((u >> 16) & 1u);               // round-to-nearest-even
  unsigned short s = (unsigned short)(u >> 16);
  return __builtin_bit_cast(bf16, s);
}

enum { ACT_NONE = 0, ACT_RELU = 1, ACT_GELU = 2, ACT_TANH = 3, ACT_SIG = 4 };

__device__ __forceinline__ float act_apply(float x, int act) {
  switch (act) {
    case ACT_RELU: return x > 0.f ? x : 0.f;
    case ACT_GELU: return 0.5f * x * (1.0f + erff(x * 0.70710678118654752f));
    case ACT_TANH: return tanhf(x);
    case ACT_SIG:  return 1.0f / (1.0f + expf(-x));
  }
  return x;
}

// --------------------------- generic WMMA GEMM ------------------------------
// Out[r,j] = act( sum_k A[r,k]*W[k,j] + bias[j] ) (+ res[remap(r), j])
// A element addr: seg[k/nseg_len] + bz*a_batch + ((r/t_dst)*t_src + r%t_dst
//                 + t_shift)*row_stride + (k%nseg_len)*k_stride
struct GemmP {
  const float* seg[9];
  const float* W;
  const float* bias;
  const float* res;
  float*       out;
  long long a_batch, w_batch, res_batch, out_batch;
  int nseg_len, k_stride, row_stride;
  int t_dst, t_src, t_shift;
  int ldw, ldo;
  int res_t_src, res_shift, res_stride;
  int M, K, Nc, act, a_act;
};

// Block tile: 32 (M) x 64 (N), 4 waves; each wave = one 16x16 A-frag row-half
// and two 16-col B groups -> two WMMAs per K-tile per wave.
// LDS is fragment-order: [slot][16 halves], slot stride 24 halves (48B).
// Fragment e-map (16-bit A 16x32): e<8 <-> k=kba+e ; e>=8 <-> k=kba+16+(e-8),
// kba = (lane<16)?0:8 -> each e-half of a slot is a contiguous 8-wide k run.
#define LDS_STRIDE 24

__launch_bounds__(128)
__global__ void gemm_wmma(GemmP p) {
  __shared__ bf16 As2[2 * 32 * LDS_STRIDE];   // [rowhalf*32+lane][16]
  __shared__ bf16 Bs2[4 * 32 * LDS_STRIDE];   // [colgroup*32+lane][16]
  const int tid  = threadIdx.x;
  const int lane = tid & 31;
  const int wave = tid >> 5;
  const int wm   = wave >> 1;            // A row-half (0/1)
  const int wn2  = (wave & 1) * 2;       // first of two B col groups
  const int m0   = blockIdx.y * 32;
  const int n0   = blockIdx.x * 64;
  const int bz   = blockIdx.z;
  const long long abz = p.a_batch * bz;
  const float* Wp = p.W + p.w_batch * bz;

  // ---- A staging: thread -> (slot 0..63, e-half); 8 consecutive k each ----
  const int slotA = tid >> 1;
  const int halfA = tid & 1;
  const int rhA   = slotA >> 5;
  const int lnA   = slotA & 31;
  const int rowAl = rhA * 16 + (lnA & 15);
  const int kRunA = ((lnA & 16) ? 8 : 0) + halfA * 16;   // first k of the run
  const int rowA  = m0 + rowAl;
  const bool okRowA = rowA < p.M;
  long long rowOffA = 0;
  if (okRowA) {
    int q = rowA / p.t_dst, t = rowA - q * p.t_dst;
    rowOffA = abz + (long long)(q * p.t_src + t + p.t_shift) * p.row_stride;
  }
  bf16* aSlot = &As2[(rhA * 32 + lnA) * LDS_STRIDE + halfA * 8];

  // ---- B staging: thread -> one slot (16 halves, k = kBase+e+(e&8)) ----
  const int gBs   = tid >> 5;
  const int lnB   = tid & 31;
  const int colBl = gBs * 16 + (lnB & 15);
  const int kBase = (lnB & 16) ? 8 : 0;
  const int colB  = n0 + colBl;
  const bool okColB = colB < p.Nc;
  const int  colCl  = okColB ? colB : (p.Nc - 1);
  bf16* bSlot = &Bs2[(gBs * 32 + lnB) * LDS_STRIDE];

  v8f acc0 = {}, acc1 = {};

  for (int k0 = 0; k0 < p.K; k0 += 32) {
    if (k0 + 32 < p.K)   // gfx1250 global_prefetch_b8 on next B panel
      __builtin_prefetch(&Wp[(long long)(k0 + 32) * p.ldw + n0], 0, 1);
    // uniform (scalar) segment pointer + k offset for this tile
    const float* sp = p.seg[k0 / p.nseg_len];
    const long long soff = (long long)(k0 % p.nseg_len) * p.k_stride;
    const int kMax = p.K - 1 - k0;       // last valid k within this tile
    // ---- stage A: branch-free clamped loads, one 16B LDS store ----
    {
      v8bf apack;
      #pragma unroll
      for (int j = 0; j < 8; ++j) {
        int kk  = kRunA + j;
        int kcl = kk < kMax ? kk : kMax;
        float v = sp[rowOffA + soff + (long long)kcl * p.k_stride];
        v = (okRowA && kk <= kMax) ? v : 0.f;
        if (p.a_act && v < 0.f) v = 0.f;
        apack[j] = f2bf(v);
      }
      *(v8bf*)aSlot = apack;
    }
    // ---- stage B: branch-free clamped loads, one 32B LDS store ----
    {
      v16bf bpack;
      #pragma unroll
      for (int e = 0; e < 16; ++e) {
        int kk  = kBase + e + (e & 8);
        int kcl = kk < kMax ? kk : kMax;
        float v = Wp[(long long)(k0 + kcl) * p.ldw + colCl];
        v = (okColB && kk <= kMax) ? v : 0.f;
        bpack[e] = f2bf(v);
      }
      *(v16bf*)bSlot = bpack;
    }
    __syncthreads();
    // ---- fragment loads: contiguous 32B per lane (2x ds_load_b128) ----
    v16bf av = *(const v16bf*)&As2[(wm * 32 + lane) * LDS_STRIDE];
    v16bf b0 = *(const v16bf*)&Bs2[((wn2 + 0) * 32 + lane) * LDS_STRIDE];
    v16bf b1 = *(const v16bf*)&Bs2[((wn2 + 1) * 32 + lane) * LDS_STRIDE];
    acc0 = __builtin_amdgcn_wmma_f32_16x16x32_bf16(false, av, false, b0,
                                                   (short)0, acc0, false, false);
    acc1 = __builtin_amdgcn_wmma_f32_16x16x32_bf16(false, av, false, b1,
                                                   (short)0, acc1, false, false);
    __syncthreads();
  }

  // ---- epilogue: C/D layout -> VGPR i = row i (lanes 0-15) / i+8 (16-31) ----
  #pragma unroll
  for (int j = 0; j < 2; ++j) {
    const v8f& acc = j ? acc1 : acc0;
    int col = n0 + (wn2 + j) * 16 + (lane & 15);
    if (col >= p.Nc) continue;
    float bvv = p.bias ? p.bias[col] : 0.f;
    #pragma unroll
    for (int i = 0; i < 8; ++i) {
      int m = m0 + wm * 16 + i + ((lane < 16) ? 0 : 8);
      if (m >= p.M) continue;
      float v = act_apply(acc[i] + bvv, p.act);
      if (p.res) {
        int q = m / p.t_dst, t = m - q * p.t_dst;
        long long ra = (long long)(q * p.res_t_src + t + p.res_shift) * p.res_stride;
        v += p.res[p.res_batch * bz + ra + col];
      }
      p.out[p.out_batch * bz + (long long)m * p.ldo + col] = v;
    }
  }
}

// --------------------------- graph generation -------------------------------

__global__ void gg_dft(const float* x, float* xf, float* fss, int total) {
  int i = blockIdx.x * blockDim.x + threadIdx.x;
  if (i >= total) return;
  int b = i / kN, n = i % kN;
  float xs[kT];
  #pragma unroll
  for (int t = 0; t < kT; ++t)
    xs[t] = x[(((long long)b * kT + t) * kN + n) * kCIN + 0];
  float mag[7]; float ss = 0.f;
  for (int k = 0; k < 7; ++k) {
    float re = 0.f, im = 0.f;
    for (int t = 0; t < kT; ++t) {
      float ang = -6.283185307179586f * (float)(k * t) / 12.0f;
      float sv, cv; sincosf(ang, &sv, &cv);
      re += xs[t] * cv; im += xs[t] * sv;
    }
    float m = sqrtf(re * re + im * im);
    mag[k] = m; ss += m * m;
  }
  float inv = 1.f / fmaxf(sqrtf(ss), 1e-12f);
  for (int k = 0; k < 7; ++k) {
    float v = mag[k] * inv;
    xf[((long long)b * 7 + k) * kN + n] = v;
    atomicAdd(&fss[b * 7 + k], v * v);
  }
}

__global__ void gg_x1(const float* xf, const float* fss, const float* x,
                      const float* TiD, const float* DiW, const float* E3,
                      const float* Wx, const float* Wd,
                      float* x1, float* bst, int total) {
  int i = blockIdx.x * blockDim.x + threadIdx.x;
  if (i >= total) return;
  int b = i / kN, n = i % kN;
  float xcat[4 * kE];
  float xn[7];
  for (int k = 0; k < 7; ++k)
    xn[k] = xf[((long long)b * 7 + k) * kN + n] / fmaxf(sqrtf(fss[b * 7 + k]), 1e-12f);
  for (int d = 0; d < kE; ++d) {
    float s = 0.f;
    for (int k = 0; k < 7; ++k) s += xn[k] * Wx[k * kE + d];
    xcat[d] = s;
  }
  for (int d = 0; d < kE; ++d) xcat[kE + d] = E3[(long long)n * kE + d];
  long long lo = (((long long)b * kT + (kT - 1)) * kN + n) * kCIN;
  int ti = (int)(x[lo + 1] * 288.0f); ti = ti < 0 ? 0 : (ti > 287 ? 287 : ti);
  int di = (int)(x[lo + 2]);          di = di < 0 ? 0 : (di > 6 ? 6 : di);
  for (int d = 0; d < kE; ++d) xcat[2 * kE + d] = TiD[ti * kE + d];
  for (int d = 0; d < kE; ++d) xcat[3 * kE + d] = DiW[di * kE + d];
  const float* wd = Wd + (long long)n * (4 * kE) * kE;
  float s0 = 0.f, s1 = 0.f;
  float* xo = x1 + (long long)i * kE;
  for (int e = 0; e < kE; ++e) {
    float a = 0.f;
    for (int f = 0; f < 4 * kE; ++f) a += xcat[f] * wd[f * kE + e];
    a = a > 0.f ? a : 0.f;
    xo[e] = a; s0 += a; s1 += a * a;
  }
  atomicAdd(&bst[b * 2], s0);
  atomicAdd(&bst[b * 2 + 1], s1);
}

__global__ void gg_dew(const float* x1, const float* bst, const float* Wxabs,
                       float* dew, int total) {
  int i = blockIdx.x * blockDim.x + threadIdx.x;
  if (i >= total) return;
  int b = i / kN;
  float cnt = (float)kN * (float)kE;
  float mu  = bst[b * 2] / cnt;
  float var = bst[b * 2 + 1] / cnt - mu * mu;
  float rs  = rsqrtf(var + 1e-8f);
  float xk[kE];
  const float* xi = x1 + (long long)i * kE;
  #pragma unroll
  for (int e = 0; e < kE; ++e) xk[e] = (xi[e] - mu) * rs;
  float* dp = dew + (long long)i * kE;
  for (int k = 0; k < kE; ++k) {
    float s = 0.f;
    #pragma unroll
    for (int e = 0; e < kE; ++e) s += xk[e] * Wxabs[e * kE + k];
    dp[k] = s;
  }
}

__launch_bounds__(256)
__global__ void adp_topk(const float* dew, const float* x1, float* AD) {
  __shared__ float row[kN];
  __shared__ float work[kN];
  __shared__ float keep[kN];
  __shared__ float rv[256];
  __shared__ int   ri[256];
  int bn = blockIdx.x, b = bn / kN, tid = threadIdx.x;
  float dv[kE];
  const float* dp = dew + (long long)bn * kE;
  #pragma unroll
  for (int e = 0; e < kE; ++e) dv[e] = dp[e];
  const float* x1b = x1 + (long long)b * kN * kE;
  for (int m = tid; m < kN; m += 256) {
    const float* xm = x1b + (long long)m * kE;
    float s = 0.f;
    #pragma unroll
    for (int e = 0; e < kE; ++e) s += dv[e] * xm[e];
    s = s > 0.f ? s : 0.f;
    row[m] = s; work[m] = s; keep[m] = 0.f;
  }
  __syncthreads();
  for (int it = 0; it < kTOPK; ++it) {        // iterative argmax (ties: low idx)
    float lm = -1.f; int li = kN;
    for (int m = tid; m < kN; m += 256) {
      float v = work[m];
      if (v > lm || (v == lm && m < li)) { lm = v; li = m; }
    }
    rv[tid] = lm; ri[tid] = li; __syncthreads();
    for (int s = 128; s > 0; s >>= 1) {
      if (tid < s) {
        if (rv[tid + s] > rv[tid] || (rv[tid + s] == rv[tid] && ri[tid + s] < ri[tid])) {
          rv[tid] = rv[tid + s]; ri[tid] = ri[tid + s];
        }
      }
      __syncthreads();
    }
    if (tid == 0) { keep[ri[0]] = 1.f; work[ri[0]] = -1.f; }
    __syncthreads();
  }
  float lmax = 0.f;
  for (int m = tid; m < kN; m += 256) {
    float z = keep[m] != 0.f ? row[m] : 0.f;
    work[m] = z; lmax = fmaxf(lmax, z);
  }
  rv[tid] = lmax; __syncthreads();
  for (int s = 128; s > 0; s >>= 1) { if (tid < s) rv[tid] = fmaxf(rv[tid], rv[tid + s]); __syncthreads(); }
  float gmax = rv[0]; __syncthreads();
  float lsum = 0.f;
  for (int m = tid; m < kN; m += 256) { float e = expf(work[m] - gmax); work[m] = e; lsum += e; }
  rv[tid] = lsum; __syncthreads();
  for (int s = 128; s > 0; s >>= 1) { if (tid < s) rv[tid] += rv[tid + s]; __syncthreads(); }
  float inv = 1.f / rv[0];
  float* out = AD + (long long)bn * kN;
  for (int m = tid; m < kN; m += 256) out[m] = work[m] * inv;
}

__launch_bounds__(256)
__global__ void adap_softmax(const float* E1, const float* E2, float* Adap) {
  __shared__ float row[kN];
  __shared__ float rv[256];
  int v = blockIdx.x, tid = threadIdx.x;
  float ev[kE];
  #pragma unroll
  for (int e = 0; e < kE; ++e) ev[e] = E1[(long long)v * kE + e];
  float lmax = -1e30f;
  for (int w = tid; w < kN; w += 256) {
    float s = 0.f;
    #pragma unroll
    for (int e = 0; e < kE; ++e) s += ev[e] * E2[(long long)w * kE + e];
    s = s > 0.f ? s : 0.f;
    row[w] = s; lmax = fmaxf(lmax, s);
  }
  rv[tid] = lmax; __syncthreads();
  for (int s = 128; s > 0; s >>= 1) { if (tid < s) rv[tid] = fmaxf(rv[tid], rv[tid + s]); __syncthreads(); }
  float gmax = rv[0]; __syncthreads();
  float lsum = 0.f;
  for (int w = tid; w < kN; w += 256) { float e = expf(row[w] - gmax); row[w] = e; lsum += e; }
  rv[tid] = lsum; __syncthreads();
  for (int s = 128; s > 0; s >>= 1) { if (tid < s) rv[tid] += rv[tid + s]; __syncthreads(); }
  float inv = 1.f / rv[0];
  for (int w = tid; w < kN; w += 256) Adap[(long long)v * kN + w] = row[w] * inv;
}

// ------------------------------- misc kernels --------------------------------

__global__ void start_kernel(const float* x, const float* sw, const float* sb,
                             float* h, int total) {
  int i = blockIdx.x * blockDim.x + threadIdx.x;
  if (i >= total) return;
  int c = i & 63;
  int rt = i >> 6;
  int t = rt % kT0;
  int bn = rt / kT0;
  int n = bn % kN, b = bn / kN;
  float acc = sb[c];
  if (t > 0) {
    const float* xp = x + (((long long)b * kT + (t - 1)) * kN + n) * kCIN;
    #pragma unroll
    for (int j = 0; j < kCIN; ++j) acc += sw[j * kC + c] * xp[j];
  }
  h[i] = acc;
}

__launch_bounds__(128)
__global__ void attn_kernel(const float* q, const float* k, const float* v,
                            float* o, int Tc) {
  __shared__ float qs[kT0 * kC], ks[kT0 * kC], vs[kT0 * kC];
  int bn = blockIdx.x;
  int tot = Tc * kC;
  const float* qb = q + (long long)bn * tot;
  const float* kb = k + (long long)bn * tot;
  const float* vb = v + (long long)bn * tot;
  for (int i = threadIdx.x; i < tot; i += blockDim.x) {
    qs[i] = qb[i]; ks[i] = kb[i]; vs[i] = vb[i];
  }
  __syncthreads();
  int h = threadIdx.x >> 4;      // 0..7 heads
  int t = threadIdx.x & 15;
  if (t >= Tc) return;
  float att[kT0];
  float mx = -1e30f;
  for (int s = 0; s <= t; ++s) {
    float d = 0.f;
    #pragma unroll
    for (int dd = 0; dd < kHD; ++dd)
      d += qs[t * kC + h * kHD + dd] * ks[s * kC + h * kHD + dd];
    d *= 0.35355339059327373f;   // 1/sqrt(HD)
    att[s] = d; mx = fmaxf(mx, d);
  }
  float den = 0.f;
  for (int s = 0; s <= t; ++s) { att[s] = expf(att[s] - mx); den += att[s]; }
  float inv = 1.f / den;
  float ov[kHD];
  #pragma unroll
  for (int dd = 0; dd < kHD; ++dd) ov[dd] = 0.f;
  for (int s = 0; s <= t; ++s) {
    float p = att[s] * inv;
    #pragma unroll
    for (int dd = 0; dd < kHD; ++dd) ov[dd] += p * vs[s * kC + h * kHD + dd];
  }
  float* ob = o + (long long)bn * tot + t * kC + h * kHD;
  #pragma unroll
  for (int dd = 0; dd < kHD; ++dd) ob[dd] = ov[dd];
}

__global__ void mul_fg(const float* f, const float* g, float* o, int total) {
  int i = blockIdx.x * blockDim.x + threadIdx.x;
  if (i < total) o[i] = f[i] * g[i];
}

__launch_bounds__(256)
__global__ void bn_kernel(float* h, const float* gamma, const float* beta, int rows) {
  __shared__ float s1[256], s2[256];
  int c = blockIdx.x, tid = threadIdx.x;
  float a = 0.f, b = 0.f;
  for (int r = tid; r < rows; r += 256) {
    float v = h[(long long)r * kC + c]; a += v; b += v * v;
  }
  s1[tid] = a; s2[tid] = b; __syncthreads();
  for (int s = 128; s > 0; s >>= 1) {
    if (tid < s) { s1[tid] += s1[tid + s]; s2[tid] += s2[tid + s]; }
    __syncthreads();
  }
  float mu = s1[0] / rows;
  float var = s2[0] / rows - mu * mu;
  float rs = rsqrtf(var + 1e-5f);
  float gg = gamma[c], bb = beta[c];
  for (int r = tid; r < rows; r += 256) {
    long long idx = (long long)r * kC + c;
    h[idx] = gg * (h[idx] - mu) * rs + bb;
  }
}

__global__ void out_transpose(const float* src, float* out, int total) {
  int i = blockIdx.x * blockDim.x + threadIdx.x;
  if (i >= total) return;
  int n = i % kN;
  int t = (i / kN) % kT;
  int b = i / (kN * kT);
  out[i] = src[((long long)b * kN + n) * kT + t];
}

// ------------------------------- host side -----------------------------------

static GemmP make_gemm(const float* A, int row_stride, int seg_len,
                       int t_dst, int t_src, int t_shift,
                       const float* W, int ldw, const float* bias,
                       float* out, int ldo, int M, int K, int Nc, int act) {
  GemmP p{};
  p.seg[0] = A; p.nseg_len = seg_len; p.k_stride = 1; p.row_stride = row_stride;
  p.t_dst = t_dst; p.t_src = t_src; p.t_shift = t_shift;
  p.W = W; p.ldw = ldw; p.bias = bias;
  p.out = out; p.ldo = ldo;
  p.M = M; p.K = K; p.Nc = Nc; p.act = act;
  return p;
}

static void launch_gemm(const GemmP& p, hipStream_t s, int Z = 1) {
  dim3 g((p.Nc + 63) / 64, (p.M + 31) / 32, Z);
  gemm_wmma<<<g, dim3(128), 0, s>>>(p);
}

extern "C" void kernel_launch(void* const* d_in, const int* in_sizes, int n_in,
                              void* d_out, int out_size, void* d_ws, size_t ws_size,
                              hipStream_t stream) {
  (void)in_sizes; (void)n_in; (void)out_size; (void)ws_size;
  const float* x       = (const float*)d_in[0];
  const float* sup0    = (const float*)d_in[1];
  const float* sup1    = (const float*)d_in[2];
  const float* TiD     = (const float*)d_in[3];
  const float* DiW     = (const float*)d_in[4];
  const float* E1      = (const float*)d_in[5];
  const float* E2      = (const float*)d_in[6];
  const float* E3      = (const float*)d_in[7];
  const float* Wx      = (const float*)d_in[8];
  const float* Wd      = (const float*)d_in[9];
  const float* Wxabs   = (const float*)d_in[10];
  const float* start_w = (const float*)d_in[11];
  const float* start_b = (const float*)d_in[12];
  const float* Wq      = (const float*)d_in[13];
  const float* bq      = (const float*)d_in[14];
  const float* Wk      = (const float*)d_in[15];
  const float* bk      = (const float*)d_in[16];
  const float* Wv      = (const float*)d_in[17];
  const float* bv      = (const float*)d_in[18];
  const float* W1      = (const float*)d_in[19];
  const float* b1      = (const float*)d_in[20];
  const float* W2      = (const float*)d_in[21];
  const float* b2      = (const float*)d_in[22];
  const float* filt_w  = (const float*)d_in[23];
  const float* filt_b  = (const float*)d_in[24];
  const float* gate_w  = (const float*)d_in[25];
  const float* gate_b  = (const float*)d_in[26];
  const float* skip_w  = (const float*)d_in[27];
  const float* skip_b  = (const float*)d_in[28];
  const float* gcn_w   = (const float*)d_in[29];
  const float* gcn_b   = (const float*)d_in[30];
  const float* bn_g    = (const float*)d_in[31];
  const float* bn_b    = (const float*)d_in[32];
  const float* e1w     = (const float*)d_in[33];
  const float* e1b     = (const float*)d_in[34];
  const float* e2w     = (const float*)d_in[35];
  const float* e2b     = (const float*)d_in[36];

  float* ws = (float*)d_ws;
  size_t off = 0;
  auto alloc = [&](size_t n) { float* p = ws + off; off += n; return p; };
  const size_t HU = (size_t)RMAX * kC;
  float* bufX  = alloc(HU);                       // h ping
  float* bufZ  = alloc(HU);                       // h pong
  float* bQ    = alloc(HU);
  float* bK    = alloc(HU);
  float* bV    = alloc(HU);
  float* bO    = alloc(HU);
  float* bY    = alloc(HU);
  float* bF1   = alloc((size_t)RMAX * 256);       // FFN mid / 4 diffusion units
  float* bS1   = alloc((size_t)kB * kN * 12 * 256);
  float* bS2   = alloc((size_t)kB * kN * 12 * 256);
  float* bAD   = alloc((size_t)kB * kN * kN);
  float* bAdap = alloc((size_t)kN * kN);
  float* bxf   = alloc((size_t)kB * 7 * kN);
  float* bx1   = alloc((size_t)kB * kN * kE);
  float* bdew  = alloc((size_t)kB * kN * kE);
  float* bacc  = alloc(80);
  float* fss = bacc; float* bst = bacc + 56;

  const int tot_bn = kB * kN;
  hipMemsetAsync(bacc, 0, 80 * sizeof(float), stream);
  gg_dft<<<(tot_bn + 255) / 256, 256, 0, stream>>>(x, bxf, fss, tot_bn);
  gg_x1 <<<(tot_bn + 255) / 256, 256, 0, stream>>>(bxf, fss, x, TiD, DiW, E3, Wx, Wd,
                                                   bx1, bst, tot_bn);
  gg_dew<<<(tot_bn + 255) / 256, 256, 0, stream>>>(bx1, bst, Wxabs, bdew, tot_bn);
  adp_topk<<<tot_bn, 256, 0, stream>>>(bdew, bx1, bAD);
  adap_softmax<<<kN, 256, 0, stream>>>(E1, E2, bAdap);

  const int tot0 = kB * kN * kT0 * kC;
  start_kernel<<<(tot0 + 255) / 256, 256, 0, stream>>>(x, start_w, start_b, bufX, tot0);

  float* h = bufX; float* hN = bufZ;
  float* sCur = bS1; float* sNext = bS2;
  int Tc = kT0, Tprev = 0;
  const int DIL[kL] = {1, 2, 1, 2, 1, 2, 1, 2};

  for (int l = 0; l < kL; ++l) {
    const int d = DIL[l], Tn = Tc - d;
    const int Mr = kB * kN * Tc;
    // ---- q / k / v projections (gelu) ----
    {
      GemmP p = make_gemm(h, kC, kC, Tc, Tc, 0, Wq + l * kC * kC, kC,
                          bq + l * kC, bQ, kC, Mr, kC, kC, ACT_GELU);
      launch_gemm(p, stream);
      p.W = Wk + l * kC * kC; p.bias = bk + l * kC; p.out = bK; launch_gemm(p, stream);
      p.W = Wv + l * kC * kC; p.bias = bv + l * kC; p.out = bV; launch_gemm(p, stream);
    }
    attn_kernel<<<tot_bn, 128, 0, stream>>>(bQ, bK, bV, bO, Tc);
    // ---- FFN 64 -> 256 -> 64 (+res) ----
    {
      GemmP p = make_gemm(bO, kC, kC, Tc, Tc, 0, W1 + l * kC * 256, 256,
                          b1 + l * 256, bF1, 256, Mr, kC, 256, ACT_GELU);
      launch_gemm(p, stream);
    }
    {
      GemmP p = make_gemm(bF1, 256, 256, Tc, Tc, 0, W2 + l * 256 * kC, kC,
                          b2 + l * kC, bY, kC, Mr, 256, kC, ACT_NONE);
      p.res = h; p.res_t_src = Tc; p.res_shift = 0; p.res_stride = kC;
      launch_gemm(p, stream);
    }
    // ---- gated dilated conv: two taps folded into K=128 GEMM ----
    const int Ms = kB * kN * Tn;
    {
      GemmP p = make_gemm(bY, kC, kC, Tn, Tc, 0, filt_w + l * 2 * kC * kC, kC,
                          filt_b + l * kC, bQ, kC, Ms, 2 * kC, kC, ACT_TANH);
      p.seg[1] = bY + d * kC;
      launch_gemm(p, stream);
      p.W = gate_w + l * 2 * kC * kC; p.bias = gate_b + l * kC;
      p.out = bK; p.act = ACT_SIG;
      launch_gemm(p, stream);
    }
    mul_fg<<<(Ms * kC + 255) / 256, 256, 0, stream>>>(bQ, bK, bV, Ms * kC);
    // ---- skip 64 -> 256 (+shifted accumulate) ----
    {
      GemmP p = make_gemm(bV, kC, kC, Tn, Tn, 0, skip_w + l * kC * 256, 256,
                          skip_b + l * 256, sNext, 256, Ms, kC, 256, ACT_NONE);
      if (l > 0) { p.res = sCur; p.res_t_src = Tprev; p.res_shift = Tprev - Tn; p.res_stride = 256; }
      launch_gemm(p, stream);
      float* t = sCur; sCur = sNext; sNext = t;
    }
    // ---- graph diffusion: 4 supports x 2 hops, A @ X with X = (N, Tn*64) ----
    const int F = Tn * kC;
    const long long WB = (long long)kN * F;
    float* Dst[8] = { bO, bY, bF1, bF1 + HU, bF1 + 2 * HU, bF1 + 3 * HU, bQ, bK };
    auto diff = [&](const float* adj, int rs, int ks, long long ab,
                    const float* Xf, float* Dout) {
      GemmP p{};
      p.seg[0] = adj; p.nseg_len = kN; p.k_stride = ks; p.row_stride = rs;
      p.t_dst = 1; p.t_src = 1; p.t_shift = 0; p.a_batch = ab;
      p.W = Xf; p.ldw = F; p.w_batch = WB;
      p.out = Dout; p.ldo = F; p.out_batch = WB;
      p.M = kN; p.K = kN; p.Nc = F; p.act = ACT_NONE;
      launch_gemm(p, stream, kB);
    };
    diff(sup0,  1, kN, 0, bV,     Dst[0]);           // 2D supports used as 'vw'
    diff(sup0,  1, kN, 0, Dst[0], Dst[1]);
    diff(sup1,  1, kN, 0, bV,     Dst[2]);
    diff(sup1,  1, kN, 0, Dst[2], Dst[3]);
    diff(bAdap, 1, kN, 0, bV,     Dst[4]);
    diff(bAdap, 1, kN, 0, Dst[4], Dst[5]);
    diff(bAD, kN, 1, (long long)kN * kN, bV,     Dst[6]);   // 3D: 'bwv'
    diff(bAD, kN, 1, (long long)kN * kN, Dst[6], Dst[7]);
    // ---- GCN combine: 9-segment K=576 GEMM, + shifted residual ----
    {
      GemmP p = make_gemm(bV, kC, kC, Tn, Tn, 0, gcn_w + l * 9 * kC * kC, kC,
                          gcn_b + l * kC, hN, kC, Ms, 9 * kC, kC, ACT_NONE);
      for (int s = 0; s < 8; ++s) p.seg[1 + s] = Dst[s];
      p.res = h; p.res_t_src = Tc; p.res_shift = d; p.res_stride = kC;
      launch_gemm(p, stream);
    }
    bn_kernel<<<kC, 256, 0, stream>>>(hN, bn_g + l * kC, bn_b + l * kC, kB * kN * Tn);
    { float* t = h; h = hN; hN = t; }
    Tprev = Tn; Tc = Tn;
  }

  // ---- end MLP: relu(skip[T=1]) -> 512 (relu) -> 12 -> transpose ----
  const int Me = kB * kN;
  {
    GemmP p = make_gemm(sCur, 256, 256, 1, 1, 0, e1w, 512, e1b,
                        bF1, 512, Me, 256, 512, ACT_RELU);
    p.a_act = 1;    // relu applied to skip on load
    launch_gemm(p, stream);
  }
  {
    GemmP p = make_gemm(bF1, 512, 512, 1, 1, 0, e2w, kT, e2b,
                        bO, kT, Me, 512, kT, ACT_NONE);
    launch_gemm(p, stream);
  }
  const int tot_out = kB * kT * kN;
  out_transpose<<<(tot_out + 255) / 256, 256, 0, stream>>>(bO, (float*)d_out, tot_out);
}
